// LSTM_84499186581683
// MI455X (gfx1250) — matrix operand order repaired and checked by reading, not compiled, not run on hardware
//
#include <hip/hip_runtime.h>
#include <hip/hip_bf16.h>
#include <math.h>

// ---------------------------------------------------------------------------
// 2-layer LSTM for MI455X (gfx1250, wave32, WMMA bf16 w/ fp32 accum)
//
//   Phase A (per layer): gx = x_bf16 @ WxT_bf16   (big WMMA GEMM, fp32 out)
//   Phase B (per layer): persistent scan kernel, 32 WGs, grid barrier/step.
//     - Each WG owns 16 hidden columns j and gate columns {q*H+j}, q=0..3.
//     - Each wave keeps its 16 k-tiles of Wh^T in VGPRs for all 2048 steps.
//     - h_prev staged into LDS each step via GLOBAL_LOAD_ASYNC_TO_LDS_B128
//       (ASYNCcnt path), gates exchanged via LDS.
// ---------------------------------------------------------------------------

typedef __attribute__((ext_vector_type(16))) __bf16 v16bf;
typedef __attribute__((ext_vector_type(8)))  __bf16 v8bf;
typedef __attribute__((ext_vector_type(8)))  float  v8f;
typedef __attribute__((ext_vector_type(4)))  int    v4i;

#define DEVI static __device__ __forceinline__

constexpr int T   = 2048;
constexpr int NB  = 64;      // batch
constexpr int D   = 512;
constexpr int H   = 512;
constexpr int G   = 4 * H;   // 2048 gate columns
constexpr int NWG = 32;      // scan workgroups (H/16)

DEVI v16bf mk16(v8bf lo, v8bf hi) {
  v16bf r;
#pragma unroll
  for (int i = 0; i < 8; ++i) { r[i] = lo[i]; r[i + 8] = hi[i]; }
  return r;
}

DEVI v8f wmma_bf16(v16bf a, v16bf b, v8f c) {
  // D = A(16x32 bf16) * B(32x16 bf16) + C(16x16 f32)
  return __builtin_amdgcn_wmma_f32_16x16x32_bf16(false, a, false, b,
                                                 (short)0, c, false, false);
}

DEVI float sigmoid_(float x) { return 1.0f / (1.0f + __expf(-x)); }

// --------------------------- prep kernels ----------------------------------

__global__ void k_f32_to_bf16(const float* __restrict__ src,
                              __bf16* __restrict__ dst, long long n) {
  long long i = (long long)blockIdx.x * blockDim.x + threadIdx.x;
  if (i < n) dst[i] = (__bf16)src[i];
}

// W[R][C] fp32  ->  WT[C][R] bf16   (R = 4H rows, C = K)
__global__ void k_transpose_to_bf16(const float* __restrict__ W,
                                    __bf16* __restrict__ WT, int R, int C) {
  long long o = (long long)blockIdx.x * blockDim.x + threadIdx.x;
  if (o >= (long long)R * C) return;
  int k = (int)(o / R);
  int g = (int)(o % R);
  WT[o] = (__bf16)W[(long long)g * C + k];
}

__global__ void k_init(__bf16* __restrict__ hbuf, unsigned* __restrict__ cnt) {
  int i = threadIdx.x + blockIdx.x * blockDim.x;
  const int n = 2 * NB * H;
  for (int j = i; j < n; j += blockDim.x * gridDim.x) hbuf[j] = (__bf16)0.0f;
  if (i == 0) *cnt = 0u;
}

// --------------------- Phase A: gx = A @ WT (bf16 -> f32) ------------------
// A: [M][512] bf16 row-major, WT: [512][G] bf16, Cout: [M][G] f32.
// Block tile 64x256, 8 waves, each wave 2 row-tiles x 4 col-tiles.

__global__ void __launch_bounds__(256)
k_gemm_gates(const __bf16* __restrict__ A, const __bf16* __restrict__ Bw,
             float* __restrict__ Cout, int M) {
  const int lane = threadIdx.x & 31;
  const int wave = threadIdx.x >> 5;
  const int wr = wave >> 2;                     // 0..1
  const int wc = wave & 3;                      // 0..3
  const long long row0 = (long long)blockIdx.y * 64;
  const int col0 = blockIdx.x * 256 + wc * 64;
  const int m = lane & 15, hf = lane >> 4;

  v8f acc[2][4] = {};

#pragma unroll
  for (int kt = 0; kt < 16; ++kt) {             // K = 512 = 16 * 32
    const int kb = kt * 32;
    v16bf bf[4];
#pragma unroll
    for (int ct = 0; ct < 4; ++ct) {
      const __bf16* bp = Bw + (long long)(kb + hf * 8) * G + col0 + ct * 16 + m;
#pragma unroll
      for (int s = 0; s < 16; ++s) bf[ct][s] = bp[(long long)((s & 7) + (s >> 3) * 16) * G];
    }
#pragma unroll
    for (int rt = 0; rt < 2; ++rt) {
      const __bf16* ap =
          A + (row0 + (long long)(wr + rt * 2) * 16 + m) * 512 + kb + hf * 8;
      v16bf af = mk16(*(const v8bf*)ap, *(const v8bf*)(ap + 16));
#pragma unroll
      for (int ct = 0; ct < 4; ++ct)
        acc[rt][ct] = wmma_bf16(af, bf[ct], acc[rt][ct]);
    }
  }

  const int n = lane & 15;
#pragma unroll
  for (int rt = 0; rt < 2; ++rt)
#pragma unroll
    for (int ct = 0; ct < 4; ++ct)
#pragma unroll
      for (int r = 0; r < 8; ++r) {
        long long row = row0 + (wr + rt * 2) * 16 + r + 8 * hf;
        Cout[row * G + col0 + ct * 16 + n] = acc[rt][ct][r];
      }
}

// --------------------- Phase B: persistent LSTM scan -----------------------

__global__ void __launch_bounds__(256)
k_lstm_scan(const float* __restrict__ gx,    // [T][NB][G] precomputed x-gates
            const __bf16* __restrict__ WhT,  // [H][G]
            __bf16* __restrict__ hbuf,       // [2][NB][H] double buffer
            unsigned* __restrict__ cnt,      // grid barrier counter
            __bf16* __restrict__ ysbf,       // [T][NB][H] bf16 hidden out
            float* __restrict__ outf) {      // [T][NB][H] f32 out or null
  __shared__ union SM {
    __bf16 h[NB * H];       // 64 KB staged h_prev (used during GEMM)
    float  gates[NB * 64];  // 16 KB gate tile [b][q*16+jj] (used after)
  } sm;

  const int tid  = threadIdx.x;
  const int lane = tid & 31;
  const int wave = tid >> 5;                  // 8 waves
  const int q    = wave & 3;                  // gate quarter (i,f,g,o)
  const int rt0  = wave >> 2;                 // row tiles {rt0, rt0+2}
  const int jbase = blockIdx.x * 16;          // owned hidden columns
  const int col0 = q * H + jbase;
  const int m = lane & 15, hf = lane >> 4;

  // Preload this wave's Wh^T fragments: 16 k-tiles, resident in VGPRs.
  v16bf Bf[16];
#pragma unroll
  for (int kt = 0; kt < 16; ++kt) {
    const __bf16* bp = WhT + (long long)(kt * 32 + hf * 8) * G + col0 + m;
#pragma unroll
    for (int s = 0; s < 16; ++s) Bf[kt][s] = bp[(long long)((s & 7) + (s >> 3) * 16) * G];
  }

  // Cell state: thread owns (b = tid>>2, jj = (tid&3)*4 + u), fixed for all t.
  float creg[4] = {0.f, 0.f, 0.f, 0.f};
  const int cb  = tid >> 2;
  const int cj0 = (tid & 3) * 4;

  for (int t = 0; t < T; ++t) {
    const __bf16* hprev = hbuf + (t & 1) * (NB * H);
    __bf16* hnext       = hbuf + ((t + 1) & 1) * (NB * H);

    // Stage h_prev (64 KB bf16) into LDS.
#if __has_builtin(__builtin_amdgcn_global_load_async_to_lds_b128) && \
    __has_builtin(__builtin_amdgcn_s_wait_asynccnt)
    {
      // Direct memory->LDS async path (ASYNCcnt), no VGPR round-trip.
      char* gp = (char*)hprev;
      __attribute__((address_space(3))) char* lp =
          (__attribute__((address_space(3))) char*)sm.h;
#pragma unroll
      for (int i = 0; i < (NB * H * 2) / (256 * 16); ++i) {
        const int off = (tid + 256 * i) * 16;
        __builtin_amdgcn_global_load_async_to_lds_b128(
            (__attribute__((address_space(1))) v4i*)(gp + off),
            (__attribute__((address_space(3))) v4i*)(lp + off),
            /*offset=*/0, /*cpol=*/0);
      }
      __builtin_amdgcn_s_wait_asynccnt(0);
    }
#else
    {
      const v8bf* s = (const v8bf*)hprev;
      v8bf* d = (v8bf*)sm.h;
#pragma unroll
      for (int i = 0; i < (NB * H / 8) / 256; ++i)
        d[tid + 256 * i] = s[tid + 256 * i];
    }
#endif
    __syncthreads();

    // gates_tile = h_prev @ WhT[:, col0:col0+16]  (two 16x16 output tiles)
    v8f acc0 = {}, acc1 = {};
#pragma unroll
    for (int kt = 0; kt < 16; ++kt) {
      const int kb = kt * 32 + hf * 8;
      const __bf16* a0 = sm.h + (rt0 * 16 + m) * H + kb;
      v16bf A0 = mk16(*(const v8bf*)a0, *(const v8bf*)(a0 + 16));
      acc0 = wmma_bf16(A0, Bf[kt], acc0);
      const __bf16* a1 = sm.h + ((rt0 + 2) * 16 + m) * H + kb;
      v16bf A1 = mk16(*(const v8bf*)a1, *(const v8bf*)(a1 + 16));
      acc1 = wmma_bf16(A1, Bf[kt], acc1);
    }
    __syncthreads();  // everyone done reading sm.h

    // Deposit gate tiles to LDS: gates[b][q*16 + n]
    const int n = lane & 15;
#pragma unroll
    for (int r = 0; r < 8; ++r) {
      sm.gates[(rt0 * 16 + r + 8 * hf) * 64 + q * 16 + n]       = acc0[r];
      sm.gates[((rt0 + 2) * 16 + r + 8 * hf) * 64 + q * 16 + n] = acc1[r];
    }
    __syncthreads();

    // LSTM cell for this thread's 4 (b, j) elements.
    const float* gxr = gx + ((long long)t * NB + cb) * G;
    if (t + 1 < T)
      __builtin_prefetch(gx + ((long long)(t + 1) * NB + cb) * G + jbase, 0, 0);
#pragma unroll
    for (int u = 0; u < 4; ++u) {
      const int jj  = cj0 + u;
      const int col = jbase + jj;
      float iv = sm.gates[cb * 64 +      jj] + gxr[0 * H + col];
      float fv = sm.gates[cb * 64 + 16 + jj] + gxr[1 * H + col];
      float gv = sm.gates[cb * 64 + 32 + jj] + gxr[2 * H + col];
      float ov = sm.gates[cb * 64 + 48 + jj] + gxr[3 * H + col];
      iv = sigmoid_(iv);
      fv = sigmoid_(fv);
      gv = tanhf(gv);
      ov = sigmoid_(ov);
      float cn = fv * creg[u] + iv * gv;
      creg[u] = cn;
      float hn = ov * tanhf(cn);
      hnext[cb * H + col] = (__bf16)hn;
      long long oidx = ((long long)t * NB + cb) * H + col;
      ysbf[oidx] = (__bf16)hn;
      if (outf) outf[oidx] = hn;
    }

    // Grid-wide barrier (monotonic counter; double-buffered h tolerates
    // 1-step skew). Release h writes, then acquire remote writes.
    __syncthreads();
    __threadfence();
    if (tid == 0) {
      atomicAdd(cnt, 1u);
      const unsigned target = (unsigned)(NWG * (t + 1));
      while (__hip_atomic_load(cnt, __ATOMIC_ACQUIRE,
                               __HIP_MEMORY_SCOPE_AGENT) < target)
        __builtin_amdgcn_s_sleep(1);
    }
    __syncthreads();
    __threadfence();
  }
}

// ------------------------------ launcher -----------------------------------

extern "C" void kernel_launch(void* const* d_in, const int* in_sizes, int n_in,
                              void* d_out, int out_size, void* d_ws,
                              size_t ws_size, hipStream_t stream) {
  (void)in_sizes; (void)n_in; (void)out_size; (void)ws_size;
  const float* x   = (const float*)d_in[0];
  const float* Wx0 = (const float*)d_in[1];
  const float* Wh0 = (const float*)d_in[2];
  const float* Wx1 = (const float*)d_in[3];
  const float* Wh1 = (const float*)d_in[4];
  float* out = (float*)d_out;

  char* ws = (char*)d_ws;
  constexpr size_t SZ_XBF  = (size_t)T * NB * D * 2;   // 128 MB
  constexpr size_t SZ_YSBF = (size_t)T * NB * H * 2;   // 128 MB
  constexpr size_t SZ_GX   = (size_t)T * NB * G * 4;   // 1 GB
  constexpr size_t SZ_HBUF = (size_t)2 * NB * H * 2;

  __bf16* xbf  = (__bf16*)ws;
  __bf16* ysbf = (__bf16*)(ws + SZ_XBF);
  __bf16* WxT0 = (__bf16*)(ws + SZ_XBF + SZ_YSBF);
  __bf16* WhT0 = WxT0 + (size_t)D * G;
  __bf16* WxT1 = WhT0 + (size_t)H * G;
  __bf16* WhT1 = WxT1 + (size_t)H * G;
  float*  gx   = (float*)((char*)(WhT1 + (size_t)H * G));
  __bf16* hbuf = (__bf16*)((char*)gx + SZ_GX);
  unsigned* cnt = (unsigned*)((char*)hbuf + SZ_HBUF);

  // Weights -> transposed bf16 [K][4H]
  {
    long long nW = (long long)G * D;  // K==512 for all four weights
    unsigned grd = (unsigned)((nW + 255) / 256);
    k_transpose_to_bf16<<<grd, 256, 0, stream>>>(Wx0, WxT0, G, D);
    k_transpose_to_bf16<<<grd, 256, 0, stream>>>(Wh0, WhT0, G, H);
    k_transpose_to_bf16<<<grd, 256, 0, stream>>>(Wx1, WxT1, G, H);
    k_transpose_to_bf16<<<grd, 256, 0, stream>>>(Wh1, WhT1, G, H);
  }
  // Input -> bf16
  {
    long long n = (long long)T * NB * D;
    k_f32_to_bf16<<<(unsigned)((n + 255) / 256), 256, 0, stream>>>(x, xbf, n);
  }

  const int M = T * NB;  // 131072
  dim3 ggrd(G / 256, M / 64);

  // Layer 0
  k_gemm_gates<<<ggrd, 256, 0, stream>>>(xbf, WxT0, gx, M);
  k_init<<<64, 256, 0, stream>>>(hbuf, cnt);
  k_lstm_scan<<<NWG, 256, 0, stream>>>(gx, WhT0, hbuf, cnt, ysbf, nullptr);

  // Layer 1 (input = layer-0 hidden, already bf16 in ysbf)
  k_gemm_gates<<<ggrd, 256, 0, stream>>>(ysbf, WxT1, gx, M);
  k_init<<<64, 256, 0, stream>>>(hbuf, cnt);
  k_lstm_scan<<<NWG, 256, 0, stream>>>(gx, WhT1, hbuf, cnt, ysbf, out);
}
